// PointCloudMambaBlock_9534827397222
// MI455X (gfx1250) — matrix operand order repaired and checked by reading, not compiled, not run on hardware
//
#include <hip/hip_runtime.h>
#include <hip/hip_bf16.h>

typedef __bf16 bf16;
typedef __attribute__((ext_vector_type(16))) __bf16 v16bf;
typedef __attribute__((ext_vector_type(8)))  float   v8f;

#define Bc   32      // batch
#define Tc   1024    // sequence length (N)
#define Hc   512     // hidden / d_model
#define G3   1536    // 3*H
#define DOUT 1024

union FragU { v16bf v; uint4 q[2]; };

// ---- WMMA fragment loaders (ISA 7.12.2 layouts) ----------------------------
// A: 16x32 bf16 tile from row-major [*, ld]. lane L: row=L&15,
//    halves k+8*(L>>4)..+7 and k+8*(L>>4)+16..+23  -> two b128 loads.
__device__ __forceinline__ v16bf load_frag_a(const bf16* A, int row0, int lane,
                                             int kc, int ld) {
  int r  = row0 + (lane & 15);
  int kh = kc + ((lane >> 4) << 3);
  const bf16* p = A + (size_t)r * ld + kh;
  FragU f;
  f.q[0] = *(const uint4*)(p);
  f.q[1] = *(const uint4*)(p + 16);
  return f.v;
}
// B: 32x16 bf16 tile computing A*W^T with W row-major [N, ld].
//    lane L: col=L&15, halves k+8*(L>>4)..+7 and k+8*(L>>4)+16..+23.
__device__ __forceinline__ v16bf load_frag_b(const bf16* W, int n0, int lane,
                                             int kc, int ld) {
  int n  = n0 + (lane & 15);
  int kb = kc + ((lane >> 4) << 3);
  const bf16* p = W + (size_t)n * ld + kb;
  FragU f;
  f.q[0] = *(const uint4*)(p);
  f.q[1] = *(const uint4*)(p + 16);
  return f.v;
}

__device__ __forceinline__ v8f wmma_bf16(v16bf a, v16bf b, v8f c) {
  return __builtin_amdgcn_wmma_f32_16x16x32_bf16(false, a, false, b,
                                                 (short)0, c, false, false);
}

// ---- small utility kernels -------------------------------------------------
__global__ __launch_bounds__(256) void cvt_bf16_kernel(const float* __restrict__ s,
                                                       bf16* __restrict__ d, int n) {
  int i = blockIdx.x * 256 + threadIdx.x;
  if (i < n) d[i] = (bf16)s[i];
}

__global__ __launch_bounds__(256) void addcvt_kernel(const float* __restrict__ a,
                                                     const float* __restrict__ b,
                                                     bf16* __restrict__ d, int n) {
  int i = blockIdx.x * 256 + threadIdx.x;
  if (i < n) d[i] = (bf16)(a[i] + b[i]);
}

// x[row, d] = b_pos[d] + coords[row,:3] . W_pos[d,:3]  (stored bf16)
__global__ __launch_bounds__(256) void posenc_kernel(const float* __restrict__ coords,
                                                     const float* __restrict__ Wp,
                                                     const float* __restrict__ bp,
                                                     bf16* __restrict__ xbf) {
  size_t idx = (size_t)blockIdx.x * 256 + threadIdx.x;     // over 32768*512
  int d = (int)(idx & (Hc - 1));
  size_t row = idx >> 9;
  const float* c = coords + row * 3;
  float v = bp[d] + c[0] * Wp[d * 3 + 0] + c[1] * Wp[d * 3 + 1] + c[2] * Wp[d * 3 + 2];
  xbf[idx] = (bf16)v;
}

// ---- generic WMMA GEMM: out[M,N] = A[M,512] * W[N,512]^T + bias ------------
__global__ __launch_bounds__(256) void gemm_bf16_kernel(const bf16* __restrict__ A,
                                                        const bf16* __restrict__ W,
                                                        const float* __restrict__ bias,
                                                        float* __restrict__ out, int N) {
  __shared__ bf16 sA[16 * Hc];                 // one 16x512 A tile, shared by 8 waves
  int mt = blockIdx.x;
  int wave = threadIdx.x >> 5, lane = threadIdx.x & 31;
  const uint4* gA = (const uint4*)(A + (size_t)mt * 16 * Hc);
  uint4* sA4 = (uint4*)sA;
#pragma unroll
  for (int u = 0; u < 4; ++u) sA4[threadIdx.x * 4 + u] = gA[threadIdx.x * 4 + u];
  __syncthreads();

  int n0 = (blockIdx.y * 8 + wave) * 16;
  v8f acc = {};
#pragma unroll
  for (int kc = 0; kc < Hc; kc += 32) {
    v16bf a = load_frag_a(sA, 0, lane, kc, Hc);      // ds_load_b128 x2
    v16bf b = load_frag_b(W, n0, lane, kc, Hc);      // global b128 x2 (L2-hot)
    acc = wmma_bf16(a, b, acc);
  }
  int lhi = lane >> 4, lcol = lane & 15;
  int n = n0 + lcol;
  float bs = bias[n];
#pragma unroll
  for (int e = 0; e < 8; ++e) {
    int m = mt * 16 + e + 8 * lhi;
    out[(size_t)m * N + n] = acc[e] + bs;
  }
}

// ---- grid-wide barrier (monotonic ticket counter) --------------------------
__device__ __forceinline__ void grid_sync(unsigned* bar, unsigned nblocks) {
  __syncthreads();
  if (threadIdx.x == 0) {
    __threadfence();
    unsigned ticket = __hip_atomic_fetch_add(&bar[0], 1u, __ATOMIC_ACQ_REL,
                                             __HIP_MEMORY_SCOPE_AGENT) + 1u;
    unsigned target = ((ticket - 1u) / nblocks + 1u) * nblocks;
    while (__hip_atomic_load(&bar[0], __ATOMIC_RELAXED,
                             __HIP_MEMORY_SCOPE_AGENT) < target)
      __builtin_amdgcn_s_sleep(1);
  }
  __syncthreads();
  __threadfence();
}

__device__ __forceinline__ float sigm(float x) { return 1.0f / (1.0f + __expf(-x)); }

// ---- persistent bidirectional GRU scan -------------------------------------
// grid = 16 blocks x 256 threads + 192KB dynamic LDS. blocks 0..7 forward,
// 8..15 backward. Each block stages its 192-row slice of Whh (3 gates x 4
// j-tiles x 16 rows x 512 k, bf16) into LDS ONCE; all 1024 steps feed B
// fragments from ds_load_b128. Each wave owns (mtile, jtile) => 16x16 of h,
// runs 48 WMMAs/step, gates in registers. gx/h_prev preloaded before the WMMA
// chain; next step's gx prefetched (global_prefetch_b8).
__global__ __launch_bounds__(256) void gru_scan_kernel(const float* __restrict__ gx,   // [2][32768][1536]
                                                       const bf16*  __restrict__ whh,  // [2][1536][512]
                                                       const float* __restrict__ bhh_f,
                                                       const float* __restrict__ bhh_b,
                                                       float* __restrict__ hout,       // [2][32768][512]
                                                       float* __restrict__ hstf,       // [2][32][512]
                                                       bf16*  __restrict__ hstbf,      // [2][2][32][512]
                                                       unsigned* __restrict__ bar) {
  extern __shared__ char smem[];
  bf16* sW = (bf16*)smem;                      // [192][512] bf16 = 192 KB

  const int dir  = blockIdx.x >> 3;
  const int blk  = blockIdx.x & 7;
  const int wave = threadIdx.x >> 5;
  const int lane = threadIdx.x & 31;
  const int mtile = wave >> 2;                 // 0..1 (rows 0-15 / 16-31)
  const int jl    = wave & 3;                  // local j-tile 0..3
  const int jt    = blk * 4 + jl;              // 0..31 (16-col chunk of H)

  const float* gx_d   = gx   + (size_t)dir * Bc * Tc * G3;
  const bf16*  whh_d  = whh  + (size_t)dir * G3 * Hc;
  const float* bhh    = dir ? bhh_b : bhh_f;
  float*       hout_d = hout + (size_t)dir * Bc * Tc * Hc;
  float*       hf32   = hstf + (size_t)dir * Bc * Hc;

  // stage this block's Whh slice: for gate g, rows [g*512 + blk*64, +64)
#pragma unroll
  for (int g = 0; g < 3; ++g) {
    const uint4* src = (const uint4*)(whh_d + ((size_t)g * Hc + blk * 64) * Hc);
    uint4* dst = (uint4*)(sW + (size_t)g * 64 * Hc);
    for (int u = threadIdx.x; u < 64 * Hc / 8; u += 256)   // 4096 x uint4
      dst[u] = src[u];
  }

  // zero initial hidden state (once, by blk==0 blocks of each direction)
  if (blk == 0) {
    for (int i = threadIdx.x; i < Bc * Hc; i += 256) {
      hf32[i] = 0.0f;
      hstbf[((size_t)dir * 2 + 0) * Bc * Hc + i] = (bf16)0.0f;
      hstbf[((size_t)dir * 2 + 1) * Bc * Hc + i] = (bf16)0.0f;
    }
  }
  __syncthreads();
  grid_sync(bar, 16);

  const int lhi = lane >> 4, lcol = lane & 15;
  const int j = jt * 16 + lcol;
  const float bhr = bhh[j], bhz = bhh[Hc + j], bhn = bhh[2 * Hc + j];

  for (int t = 0; t < Tc; ++t) {
    const int p = dir ? (Tc - 1 - t) : t;
    const bf16* hA = hstbf + ((size_t)dir * 2 + (t & 1)) * Bc * Hc;        // read buf
    bf16*       hW = hstbf + ((size_t)dir * 2 + ((t + 1) & 1)) * Bc * Hc;  // write buf

    // preload gate operands for this step (overlaps with the WMMA chain)
    float gxr[8], gxz[8], gxn[8], hp[8];
#pragma unroll
    for (int e = 0; e < 8; ++e) {
      int b = mtile * 16 + e + 8 * lhi;
      const float* q = gx_d + ((size_t)b * Tc + p) * G3;
      gxr[e] = q[j];
      gxz[e] = q[Hc + j];
      gxn[e] = q[2 * Hc + j];
      hp[e]  = hf32[(size_t)b * Hc + j];
    }

    v8f cr = {}, cz = {}, cn = {};
#pragma unroll
    for (int kc = 0; kc < Hc; kc += 32) {
      v16bf a  = load_frag_a(hA, mtile * 16, lane, kc, Hc);   // global (L2-hot)
      v16bf br = load_frag_b(sW, 0 * 64 + jl * 16, lane, kc, Hc);  // ds_load
      v16bf bz = load_frag_b(sW, 1 * 64 + jl * 16, lane, kc, Hc);
      v16bf bn = load_frag_b(sW, 2 * 64 + jl * 16, lane, kc, Hc);
      cr = wmma_bf16(a, br, cr);
      cz = wmma_bf16(a, bz, cz);
      cn = wmma_bf16(a, bn, cn);
    }

#pragma unroll
    for (int e = 0; e < 8; ++e) {
      int b = mtile * 16 + e + 8 * lhi;
      float r  = sigm(gxr[e] + cr[e] + bhr);
      float z  = sigm(gxz[e] + cz[e] + bhz);
      float nn = tanhf(gxn[e] + r * (cn[e] + bhn));
      float hn = (1.0f - z) * nn + z * hp[e];
      hout_d[((size_t)b * Tc + p) * Hc + j] = hn;
      hf32[(size_t)b * Hc + j] = hn;
      hW[(size_t)b * Hc + j] = (bf16)hn;
    }

    // prefetch next step's gx rows into cache (global_prefetch_b8)
    if (t + 1 < Tc) {
      const int pn = dir ? (Tc - 2 - t) : (t + 1);
#pragma unroll
      for (int e = 0; e < 8; ++e) {
        int b = mtile * 16 + e + 8 * lhi;
        const float* q = gx_d + ((size_t)b * Tc + pn) * G3;
        __builtin_prefetch(q + j, 0, 1);
        __builtin_prefetch(q + Hc + j, 0, 1);
        __builtin_prefetch(q + 2 * Hc + j, 0, 1);
      }
    }

    grid_sync(bar, 16);  // everyone done before next step reads hW
  }
}

// ---- BatchNorm1d (channel = sequence index n) + ReLU, in place -------------
__global__ __launch_bounds__(256) void bn_relu_kernel(float* __restrict__ y,
                                                      const float* __restrict__ gamma,
                                                      const float* __restrict__ beta) {
  __shared__ float s1[256], s2[256];
  const int n = blockIdx.x;            // channel 0..1023
  const int tid = threadIdx.x;
  float sum = 0.0f, sq = 0.0f;
  for (int i = tid; i < Bc * DOUT; i += 256) {
    int b = i >> 10, d = i & (DOUT - 1);
    float v = y[((size_t)b * Tc + n) * DOUT + d];
    sum += v; sq += v * v;
  }
  s1[tid] = sum; s2[tid] = sq;
  __syncthreads();
  for (int s = 128; s > 0; s >>= 1) {
    if (tid < s) { s1[tid] += s1[tid + s]; s2[tid] += s2[tid + s]; }
    __syncthreads();
  }
  const float inv_cnt = 1.0f / (float)(Bc * DOUT);
  float mean = s1[0] * inv_cnt;
  float var  = s2[0] * inv_cnt - mean * mean;
  float scl  = gamma[n] * rsqrtf(var + 1e-5f);
  float sh   = beta[n] - mean * scl;
  for (int i = tid; i < Bc * DOUT; i += 256) {
    int b = i >> 10, d = i & (DOUT - 1);
    size_t idx = ((size_t)b * Tc + n) * DOUT + d;
    float v = y[idx] * scl + sh;
    y[idx] = v > 0.0f ? v : 0.0f;
  }
}

// ---- host-side orchestration ------------------------------------------------
extern "C" void kernel_launch(void* const* d_in, const int* in_sizes, int n_in,
                              void* d_out, int out_size, void* d_ws, size_t ws_size,
                              hipStream_t stream) {
  const float* coords = (const float*)d_in[0];
  const float* W_pos  = (const float*)d_in[1];
  const float* b_pos  = (const float*)d_in[2];
  const float* Wih_f  = (const float*)d_in[3];
  const float* Whh_f  = (const float*)d_in[4];
  const float* bih_f  = (const float*)d_in[5];
  const float* bhh_f  = (const float*)d_in[6];
  const float* Wih_b  = (const float*)d_in[7];
  const float* Whh_b  = (const float*)d_in[8];
  const float* bih_b  = (const float*)d_in[9];
  const float* bhh_b  = (const float*)d_in[10];
  const float* W_lin  = (const float*)d_in[11];
  const float* b_lin  = (const float*)d_in[12];
  const float* gamma  = (const float*)d_in[13];
  const float* beta   = (const float*)d_in[14];

  const size_t ROWS = (size_t)Bc * Tc;     // 32768
  char* p = (char*)d_ws;
  auto take = [&](size_t bytes) -> void* {
    void* r = (void*)p;
    p += (bytes + 255) & ~(size_t)255;
    return r;
  };
  unsigned* bar   = (unsigned*)take(256);
  bf16* xbf       = (bf16*)take(ROWS * Hc * 2);                 // 32 MB
  bf16* wih_bf    = (bf16*)take((size_t)2 * G3 * Hc * 2);       // 3 MB
  bf16* whh_bf    = (bf16*)take((size_t)2 * G3 * Hc * 2);       // 3 MB
  bf16* wlin_bf   = (bf16*)take((size_t)DOUT * Hc * 2);         // 1 MB
  float* gx       = (float*)take((size_t)2 * ROWS * G3 * 4);    // 384 MB
  float* hout     = (float*)take((size_t)2 * ROWS * Hc * 4);    // 128 MB
  float* hstf     = (float*)take((size_t)2 * Bc * Hc * 4);
  bf16*  hstbf    = (bf16*)take((size_t)2 * 2 * Bc * Hc * 2);
  bf16*  hsum     = (bf16*)take(ROWS * Hc * 2);                 // 32 MB

  hipMemsetAsync(bar, 0, 256, stream);

  // weights -> bf16
  cvt_bf16_kernel<<<(G3 * Hc + 255) / 256, 256, 0, stream>>>(Wih_f, wih_bf, G3 * Hc);
  cvt_bf16_kernel<<<(G3 * Hc + 255) / 256, 256, 0, stream>>>(Wih_b, wih_bf + (size_t)G3 * Hc, G3 * Hc);
  cvt_bf16_kernel<<<(G3 * Hc + 255) / 256, 256, 0, stream>>>(Whh_f, whh_bf, G3 * Hc);
  cvt_bf16_kernel<<<(G3 * Hc + 255) / 256, 256, 0, stream>>>(Whh_b, whh_bf + (size_t)G3 * Hc, G3 * Hc);
  cvt_bf16_kernel<<<(DOUT * Hc + 255) / 256, 256, 0, stream>>>(W_lin, wlin_bf, DOUT * Hc);

  // positional encoding -> x (bf16)
  posenc_kernel<<<(unsigned)(ROWS * Hc / 256), 256, 0, stream>>>(coords, W_pos, b_pos, xbf);

  // gx = x @ Wih^T + bih  (both directions; backward indexed time-reversed in scan)
  gemm_bf16_kernel<<<dim3(ROWS / 16, G3 / 128), 256, 0, stream>>>(xbf, wih_bf, bih_f, gx, G3);
  gemm_bf16_kernel<<<dim3(ROWS / 16, G3 / 128), 256, 0, stream>>>(
      xbf, wih_bf + (size_t)G3 * Hc, bih_b, gx + (size_t)ROWS * G3, G3);

  // persistent bidirectional GRU scan (16 co-resident blocks, 192KB LDS each)
  gru_scan_kernel<<<16, 256, 192 * 1024, stream>>>(gx, whh_bf, bhh_f, bhh_b,
                                                   hout, hstf, hstbf, bar);

  // h = hf + hb  (bf16 for the output GEMM)
  addcvt_kernel<<<(unsigned)(ROWS * Hc / 256), 256, 0, stream>>>(
      hout, hout + (size_t)ROWS * Hc, hsum, (int)(ROWS * Hc));

  // y = h @ W_lin^T + b_lin  -> straight into d_out
  gemm_bf16_kernel<<<dim3(ROWS / 16, DOUT / 128), 256, 0, stream>>>(
      hsum, wlin_bf, b_lin, (float*)d_out, DOUT);

  // BatchNorm over (B, D_OUT) per sequence index + ReLU, in place
  bn_relu_kernel<<<Tc, 256, 0, stream>>>((float*)d_out, gamma, beta);
}
